// CrossAttention_39762807226744
// MI455X (gfx1250) — matrix operand order repaired and checked by reading, not compiled, not run on hardware
//
#include <hip/hip_runtime.h>
#include <hip/hip_bf16.h>

typedef __attribute__((ext_vector_type(16))) _Float16 v16h;
typedef __attribute__((ext_vector_type(8)))  float    v8f;

#define WMMA_F16(a, b, c) \
  __builtin_amdgcn_wmma_f32_16x16x32_f16(false, (a), false, (b), (short)0, (c), false, false)

// ---- fragment loaders (contiguous k = 16h + s layouts, wave32) -------------------------
// A 16x32 f16: lane<16 -> row=lane,  K = 0..15 (+ko)
//              lane>=16 -> row=lane-16, K = 16..31 (+ko)
__device__ inline v16h a_frag_f16(const _Float16* __restrict__ row, int ko, int lane) {
  int o = (lane & 16) ? 16 : 0;
  union { uint4 u[2]; v16h h; } x;
  x.u[0] = *(const uint4*)(row + ko + o);
  x.u[1] = *(const uint4*)(row + ko + 8 + o);
  return x.h;
}
__device__ inline v16h a_frag_f32(const float* __restrict__ row, int ko, int lane) {
  int o = (lane & 16) ? 16 : 0;
  const float* p0 = row + ko + o;
  const float* p1 = row + ko + 8 + o;
  v16h a;
#pragma unroll
  for (int i = 0; i < 8; ++i) { a[i] = (_Float16)p0[i]; a[8 + i] = (_Float16)p1[i]; }
  return a;
}
// B 32x16 f16: lane holds column N=lane&15; lane<16 -> K=0..15, lane>=16 -> K=16..31
// caller passes pointer to the 16 contiguous halves for this lane
__device__ inline v16h b_frag_f16(const _Float16* __restrict__ p) {
  union { uint4 u[2]; v16h h; } x;
  x.u[0] = *(const uint4*)(p);
  x.u[1] = *(const uint4*)(p + 8);
  return x.h;
}
__device__ inline v16h b_frag_f32(const float* __restrict__ p) {
  v16h b;
#pragma unroll
  for (int i = 0; i < 16; ++i) b[i] = (_Float16)p[i];
  return b;
}

// ---- GEMM: C[16384x256] = X[16384x256] @ W[256x256]^T + bias ---------------
// mode 0: f32 row-major to outF
// mode 1: f16 to outH[bh][s][d]      (head-major)
// mode 2: f16 to outH[bh][d][s]      (head-major transposed, for V)
__global__ __launch_bounds__(128) void gemm_xwt(
    const float* __restrict__ X, const float* __restrict__ W,
    const float* __restrict__ bias, float* __restrict__ outF,
    _Float16* __restrict__ outH, int mode) {
  int tid = threadIdx.x, lane = tid & 31, wave = tid >> 5;
  int rowbase = blockIdx.x * 16;
  int colbase = blockIdx.y * 64 + wave * 16;
  const float* xrow = X + (size_t)(rowbase + (lane & 15)) * 256;
  const float* wrow = W + (size_t)(colbase + (lane & 15)) * 256 + ((lane & 16) ? 16 : 0);
  v8f c = {};
#pragma unroll
  for (int ko = 0; ko < 256; ko += 32) {
    v16h a = a_frag_f32(xrow, ko, lane);
    v16h b = b_frag_f32(wrow + ko);
    c = WMMA_F16(a, b, c);
  }
  int ncol = colbase + (lane & 15);
  float bv = bias[ncol];
  int mbase = rowbase + ((lane & 16) ? 8 : 0);
#pragma unroll
  for (int r = 0; r < 8; ++r) {
    float v = c[r] + bv;
    int m = mbase + r;
    if (mode == 0) {
      outF[(size_t)m * 256 + ncol] = v;
    } else {
      int bi = m >> 11, s = m & 2047, h = ncol >> 6, d = ncol & 63;
      size_t bh = (size_t)bi * 4 + h;
      size_t idx = (mode == 1) ? ((bh * 2048 + s) * 64 + d)
                               : ((bh * 64 + d) * 2048 + s);
      outH[idx] = (_Float16)v;
    }
  }
}

// ---- attention: one WG (4 waves) per (b, h, 16-row q tile) -----------------
__global__ __launch_bounds__(128) void attn_kernel(
    const _Float16* __restrict__ Q, const _Float16* __restrict__ K,
    const _Float16* __restrict__ Vt, const unsigned char* __restrict__ mask,
    float* __restrict__ attnOut, float* __restrict__ Ohead) {
  __shared__ float sc[16][2048];     // 128 KB score tile
  __shared__ float red[16][8];
  __shared__ float rowmax[16], rowinv[16], raw0[16];
  __shared__ int   flags[16];
  __shared__ float obuf[16][64];

  int tid = threadIdx.x, lane = tid & 31, wave = tid >> 5;
  int qt = blockIdx.x, h = blockIdx.y, bi = blockIdx.z;
  size_t bh = (size_t)bi * 4 + h;
  int qrow = qt * 16;
  const _Float16* qbase = Q + (bh * 2048 + (size_t)qrow) * 64;
  const _Float16* kall  = K + bh * 2048 * 64;
  const _Float16* vbase = Vt + bh * 64 * 2048;
  const unsigned char* mrow0 = mask + ((size_t)bi * 2048 + qrow) * 2048;

  for (int i = tid; i < 16; i += 128) flags[i] = 0;
  for (int i = tid; i < 16 * 64; i += 128) ((float*)obuf)[i] = 0.f;
  __syncthreads();

  // Q fragments (A operand), reused over all key tiles
  const _Float16* qr = qbase + (size_t)(lane & 15) * 64;
  v16h qa0 = a_frag_f16(qr, 0, lane);
  v16h qa1 = a_frag_f16(qr, 32, lane);

  const float scale = 0.125f;  // 1/sqrt(64)
  int mr = (lane & 16) ? 8 : 0;
  int nc = lane & 15;

  // ---- phase 1: scores + scale + mask -> LDS ----
  for (int kt = wave; kt < 128; kt += 4) {
    int kb = kt * 16;
    const _Float16* kr = kall + (size_t)(kb + nc) * 64 + ((lane & 16) ? 16 : 0);
    v8f c = {};
    c = WMMA_F16(qa0, b_frag_f16(kr), c);
    c = WMMA_F16(qa1, b_frag_f16(kr + 32), c);
#pragma unroll
    for (int r = 0; r < 8; ++r) {
      int m = mr + r;
      float raw = c[r] * scale;
      unsigned char mk = mrow0[(size_t)m * 2048 + kb + nc];
      if (mk) flags[m] = 1;
      if ((kb | nc) == 0) raw0[m] = raw;  // wave 0 owns kt==0
      sc[m][kb + nc] = mk ? raw : -1e9f;
    }
  }
  __syncthreads();
  // mask-fix: rows with no valid key get key 0 unmasked
  if (tid < 16 && flags[tid] == 0) sc[tid][0] = raw0[tid];
  __syncthreads();

  // ---- phase 2: softmax stats (8 threads per row) ----
  int row = tid >> 3, part = tid & 7;
  float* srow = &sc[row][part * 256];
  float mx = -3.0e38f;
  for (int i = 0; i < 256; ++i) mx = fmaxf(mx, srow[i]);
  red[row][part] = mx;
  __syncthreads();
  if (part == 0) {
    float m2 = red[row][0];
#pragma unroll
    for (int i = 1; i < 8; ++i) m2 = fmaxf(m2, red[row][i]);
    rowmax[row] = m2;
  }
  __syncthreads();
  float rm = rowmax[row];
  float sum = 0.f;
  for (int i = 0; i < 256; ++i) {
    float e = __expf(srow[i] - rm);
    srow[i] = e;           // keep unnormalized p in LDS
    sum += e;
  }
  red[row][part] = sum;
  __syncthreads();
  if (part == 0) {
    float s2 = 0.f;
#pragma unroll
    for (int i = 0; i < 8; ++i) s2 += red[row][i];
    rowinv[row] = 1.f / s2;  // >= 1 always (max element contributes exp(0)=1)
  }
  __syncthreads();

  // ---- phase 3a: write normalized attn (single pass over HBM buffer) ----
  float inv = rowinv[row];
  float* arow = attnOut + ((bh * 2048) + (size_t)(qrow + row)) * 2048 + part * 256;
  for (int i = 0; i < 256; i += 4) {
    float4 v = make_float4(srow[i] * inv, srow[i + 1] * inv,
                           srow[i + 2] * inv, srow[i + 3] * inv);
    *(float4*)(arow + i) = v;
  }

  // ---- phase 3b: O = P @ V via WMMA (B from transposed V) ----
  v8f oacc0 = {}, oacc1 = {}, oacc2 = {}, oacc3 = {};
  int am = lane & 15;
  float pinv = rowinv[am];
  int ao = (lane & 16) ? 16 : 0;
  for (int kt = wave; kt < 64; kt += 4) {
    int kb = kt * 32;
    const float* s0 = &sc[am][kb + ao];
    const float* s1 = &sc[am][kb + 8 + ao];
    v16h pa;
#pragma unroll
    for (int i = 0; i < 8; ++i) {
      pa[i]     = (_Float16)(s0[i] * pinv);
      pa[8 + i] = (_Float16)(s1[i] * pinv);
    }
    const _Float16* vr = vbase + (size_t)nc * 2048 + kb + ((lane & 16) ? 16 : 0);
    oacc0 = WMMA_F16(pa, b_frag_f16(vr), oacc0);
    oacc1 = WMMA_F16(pa, b_frag_f16(vr + 16 * 2048), oacc1);
    oacc2 = WMMA_F16(pa, b_frag_f16(vr + 32 * 2048), oacc2);
    oacc3 = WMMA_F16(pa, b_frag_f16(vr + 48 * 2048), oacc3);
  }
  // cross-wave reduction in LDS
#pragma unroll
  for (int r = 0; r < 8; ++r) {
    atomicAdd(&obuf[mr + r][nc],      oacc0[r]);
    atomicAdd(&obuf[mr + r][16 + nc], oacc1[r]);
    atomicAdd(&obuf[mr + r][32 + nc], oacc2[r]);
    atomicAdd(&obuf[mr + r][48 + nc], oacc3[r]);
  }
  __syncthreads();
  // emit Ohead in [B, SQ, 256] layout (plain GEMM input for Wo)
  int om = tid >> 3, od = (tid & 7) * 8;
  float* orow = Ohead + ((size_t)bi * 2048 + qrow + om) * 256 + h * 64 + od;
  *(float4*)(orow)     = *(float4*)&obuf[om][od];
  *(float4*)(orow + 4) = *(float4*)&obuf[om][od + 4];
}

extern "C" void kernel_launch(void* const* d_in, const int* in_sizes, int n_in,
                              void* d_out, int out_size, void* d_ws, size_t ws_size,
                              hipStream_t stream) {
  const float* query = (const float*)d_in[0];
  const float* key   = (const float*)d_in[1];
  const float* value = (const float*)d_in[2];
  const unsigned char* mask = (const unsigned char*)d_in[3];
  const float* Wq = (const float*)d_in[4];  const float* bq = (const float*)d_in[5];
  const float* Wk = (const float*)d_in[6];  const float* bk = (const float*)d_in[7];
  const float* Wv = (const float*)d_in[8];  const float* bv = (const float*)d_in[9];
  const float* Wo = (const float*)d_in[10]; const float* bo = (const float*)d_in[11];

  const size_t QKV = (size_t)8 * 4 * 2048 * 64;  // 4,194,304 elems per tensor
  _Float16* qf = (_Float16*)d_ws;
  _Float16* kf = qf + QKV;
  _Float16* vt = kf + QKV;
  float*    oh = (float*)(vt + QKV);             // [B, SQ, 256] f32

  float* outO  = (float*)d_out;                  // [B, SQ, 256]
  float* attnO = outO + (size_t)8 * 2048 * 256;  // [B, NH, SQ, SK]

  dim3 gg(1024, 4, 1);
  gemm_xwt<<<gg, 128, 0, stream>>>(query, Wq, bq, nullptr, qf, 1);
  gemm_xwt<<<gg, 128, 0, stream>>>(key,   Wk, bk, nullptr, kf, 1);
  gemm_xwt<<<gg, 128, 0, stream>>>(value, Wv, bv, nullptr, vt, 2);
  attn_kernel<<<dim3(128, 4, 8), 128, 0, stream>>>(qf, kf, vt, mask, attnO, oh);
  gemm_xwt<<<gg, 128, 0, stream>>>(oh, Wo, bo, outO, nullptr, 0);
}
